// FourierFilterBanks_52115133170147
// MI455X (gfx1250) — matrix operand order repaired and checked by reading, not compiled, not run on hardware
//
#include <hip/hip_runtime.h>
#include <stdint.h>

// ---------------- architecture constants ----------------
#define DLEN   136            // feature width
#define KPAD   160            // K padded to 5 x 32
#define NPAD   144            // N padded to 9 x 16
#define TPTS   128            // points per block (8 waves x 16 rows)
#define NLAYER 8
#define TBITS  19

#define SIN_W0   201.06192982974676f   // 64*pi
#define SIN_W0H  402.12385965949352f   // 128*pi
#define F19_7    2.7142857142857144f   // 19/7
#define PI_HALF  1.5707963267948966f

typedef __attribute__((ext_vector_type(16))) _Float16 v16h;
typedef __attribute__((ext_vector_type(8)))  _Float16 v8h;
typedef __attribute__((ext_vector_type(8)))  float    v8f;
typedef __attribute__((ext_vector_type(4)))  unsigned v4u;
typedef __attribute__((ext_vector_type(8)))  int      v8i;
typedef __attribute__((ext_vector_type(4)))  int      v4i;

// ---------------- LDS layout (dynamic) ----------------
#define OFF_W     0
#define SZ_W      (NPAD * KPAD * 2)          // 46080 B : f16 weights [144][160]
#define OFF_ACTA  (OFF_W + SZ_W)
#define SZ_ACT    (TPTS * KPAD * 2)          // 40960 B : f16 activations [128][160]
#define OFF_ACTB  (OFF_ACTA + SZ_ACT)
#define OFF_EMB   (OFF_ACTB + SZ_ACT)
#define OFF_GRID  (OFF_EMB + SZ_ACT)
#define SZ_GRID   (TPTS * 64 * 4)            // 32768 B : f32 grid feats [128][8][8]
#define OFF_IN    (OFF_GRID + SZ_GRID)
#define SZ_IN     (TPTS * 4 * 4)             //  2048 B : f32 inputs [128][4]
#define OFF_STAGE (OFF_IN + SZ_IN)
#define SZ_STAGE  (DLEN * DLEN * 4)          // 73984 B : f32 TDM staging
#define SMEM_BYTES (OFF_STAGE + SZ_STAGE)    // 277760 B (< 320KB WGP LDS)

#if __has_builtin(__builtin_amdgcn_tensor_load_to_lds)
#define HAVE_TDM 1
#else
#define HAVE_TDM 0
#endif

// Issue a TDM load of a [136][136] f32 matrix (row stride 136) into LDS at lds_off.
__device__ __forceinline__ void tdm_load_matrix(const float* g, unsigned lds_off) {
#if HAVE_TDM
    const unsigned long long ga = (unsigned long long)(uintptr_t)g;
    v4u g0;
    g0[0] = 1u;                                   // count=1 (valid), no gather
    g0[1] = lds_off;                              // lds_addr (bytes)
    g0[2] = (unsigned)ga;                         // global_addr[31:0]
    g0[3] = (unsigned)((ga >> 32) & 0x01FFFFFFu)  // global_addr[56:32]
          | (2u << 30);                           // type = 2 ("image")
    v8i g1;
    g1[0] = (int)(2u << 16);                      // wg_mask=0, data_size=4B
    g1[1] = (int)((unsigned)DLEN << 16);          // tensor_dim0 = 136
    g1[2] = (int)((unsigned)DLEN << 16);          // tensor_dim1 = 136
    g1[3] = (int)((unsigned)DLEN << 16);          // tile_dim0   = 136
    g1[4] = (int)DLEN;                            // tile_dim1   = 136, tile_dim2 = 0
    g1[5] = (int)DLEN;                            // tensor_dim0_stride = 136
    g1[6] = 0;
    g1[7] = 0;
    const v4i z4 = {0, 0, 0, 0};
    const v8i z8 = {0, 0, 0, 0, 0, 0, 0, 0};
    // 6-arg form on this toolchain: (g0, g1, g2, g3, g4, cpol)
    __builtin_amdgcn_tensor_load_to_lds(g0, g1, z4, z4, z8, 0);
#else
    (void)g; (void)lds_off;
#endif
}

__device__ __forceinline__ void tdm_wait() {
#if HAVE_TDM
    __builtin_amdgcn_s_wait_tensorcnt(0);
#endif
}

// posenc(grid)[j]: [v(8) | sin(v*f) (64) | cos(v*f) (64)] -- fully branchless
__device__ __forceinline__ float posenc_val(const float* gp, int j) {
    const int t0   = (j < 8) ? 0 : ((j < 72) ? (j - 8) : (j - 72));
    const int t    = t0 & 63;                       // clamp pad lanes in-bounds
    const int fidx = (j < 8) ? j : (t >> 3);
    const float g  = gp[fidx];                      // single unconditional ds_load
    const float ph = (j >= 72) ? PI_HALF : 0.0f;    // cos(x) = sin(x + pi/2)
    const float sv = __sinf(g * exp2f(F19_7 * (float)(t & 7)) + ph);
    const float v  = (j < 8) ? g : sv;
    return (j < DLEN) ? v : 0.0f;
}

// One 16x16 output tile of [16 x KPAD] @ [KPAD x 16] via 5 chained WMMAs.
__device__ __forceinline__ v8f mm_tile(const _Float16* aBase, const _Float16* sW,
                                       int n0, int lane) {
    v8f c = {};
    const int m  = lane & 15;
    const int hi = lane >> 4;
    const _Float16* arow = aBase + m * KPAD;
    const _Float16* brow = sW + (n0 + m) * KPAD;
#pragma unroll
    for (int ks = 0; ks < 5; ++ks) {
        const int k0 = ks * 32;
        union { v16h v; v8h h[2]; } a, b;
        // A 16x32 f16 layout (per lane-half K interleave)
        a.h[0] = *(const v8h*)(arow + k0 + hi * 8);
        a.h[1] = *(const v8h*)(arow + k0 + 16 + hi * 8);
        // B 32x16 f16 layout (per lane-half K interleave, same as A)
        b.h[0] = *(const v8h*)(brow + k0 + hi * 8);
        b.h[1] = *(const v8h*)(brow + k0 + 16 + hi * 8);
        c = __builtin_amdgcn_wmma_f32_16x16x32_f16(false, a.v, false, b.v,
                                                   (short)0, c, false, false);
    }
    return c;
}

__global__ void __launch_bounds__(256, 1)
FourierFilterBanks_52115133170147_kernel(const float* __restrict__ inp,
                                         const float* __restrict__ tables,
                                         const float* __restrict__ W0m,
                                         const float* __restrict__ b0,
                                         const float* __restrict__ Wf,
                                         const float* __restrict__ bfv,
                                         const float* __restrict__ Wo,
                                         const float* __restrict__ bov,
                                         float* __restrict__ out) {
    extern __shared__ __align__(16) char smem[];
    _Float16* sW    = (_Float16*)(smem + OFF_W);
    _Float16* sActA = (_Float16*)(smem + OFF_ACTA);
    _Float16* sActB = (_Float16*)(smem + OFF_ACTB);
    _Float16* sEmb  = (_Float16*)(smem + OFF_EMB);
    float*    sGrid = (float*)(smem + OFF_GRID);
    float*    sIn   = (float*)(smem + OFF_IN);
    float*    sStage = (float*)(smem + OFF_STAGE);

    const int tid    = threadIdx.x;
    const int lane   = tid & 31;
    const int wave   = tid >> 5;
    const int blockP = blockIdx.x * TPTS;
    const unsigned stageOff = (unsigned)(uintptr_t)sStage;  // LDS byte offset

    // ---- Phase 0: TDM prefetch of first weight matrix + async input copy ----
    if (wave == 0) tdm_load_matrix(Wf, stageOff);           // overlaps hash phase
    {
        int idx = tid;
#pragma unroll
        for (int it = 0; it < 2; ++it, idx += 256) {
            if (idx < TPTS * 3) {
                int p = idx / 3, c = idx - 3 * p;
                unsigned lds = (unsigned)(uintptr_t)(&sIn[p * 4 + c]);
                unsigned long long ga =
                    (unsigned long long)(uintptr_t)(inp + (size_t)(blockP + p) * 3 + c);
                asm volatile("global_load_async_to_lds_b32 %0, %1, off"
                             :: "v"(lds), "v"(ga) : "memory");
            }
        }
        // zero-init padded weight buffer while copies fly
        for (int i = tid; i < NPAD * KPAD; i += 256) sW[i] = (_Float16)0.0f;
        asm volatile("s_wait_asynccnt 0" ::: "memory");
    }
    __syncthreads();

    // ---- Phase 1a: hash-grid features (levels 0..6), 2 threads per point ----
    {
        const int p    = tid >> 1;
        const int half = tid & 1;
        const float gx = (sIn[p * 4 + 0] + 1.0f) * 0.5f;
        const float gy = (sIn[p * 4 + 1] + 1.0f) * 0.5f;
        const float gz = (sIn[p * 4 + 2] + 1.0f) * 0.5f;
        const int l0 = half ? 4 : 0;
        const int l1 = half ? 7 : 4;
        for (int lvl = l0; lvl < l1; ++lvl) {
            const float res = (float)(16 << lvl);   // growth factor is exactly 2
            const float fx = gx * res, fy = gy * res, fz = gz * res;
            const float px = floorf(fx), py = floorf(fy), pz = floorf(fz);
            const float frx = fx - px, fry = fy - py, frz = fz - pz;
            const unsigned ix = (unsigned)(int)px;
            const unsigned iy = (unsigned)(int)py;
            const unsigned iz = (unsigned)(int)pz;
            float acc[8];
#pragma unroll
            for (int q = 0; q < 8; ++q) acc[q] = 0.0f;
#pragma unroll
            for (int cr = 0; cr < 8; ++cr) {
                const unsigned c0 = ix + ((cr >> 2) & 1);
                const unsigned c1 = iy + ((cr >> 1) & 1);
                const unsigned c2 = iz + (cr & 1);
                const unsigned h  = (c0 * 1u) ^ (c1 * 2654435761u) ^ (c2 * 805459861u);
                const unsigned id = h & ((1u << TBITS) - 1u);
                const float* f = tables + (((size_t)lvl << TBITS) + id) * 8;
                const float w = ((cr & 4) ? frx : 1.0f - frx) *
                                ((cr & 2) ? fry : 1.0f - fry) *
                                ((cr & 1) ? frz : 1.0f - frz);
#pragma unroll
                for (int q = 0; q < 8; ++q) acc[q] += w * f[q];
            }
#pragma unroll
            for (int q = 0; q < 8; ++q) sGrid[p * 64 + lvl * 8 + q] = acc[q];
        }
    }

    // ---- Phase 1b: input SIREN layer -> sActA; zero sActB/sEmb pads ----
    for (int i = tid; i < TPTS * KPAD; i += 256) {
        const int p = i / KPAD, j = i - p * KPAD;
        float v = 0.0f;
        if (j < DLEN) {
            const float* wr = W0m + j * 3;
            v = __sinf(SIN_W0 * (wr[0] * sIn[p * 4 + 0] +
                                 wr[1] * sIn[p * 4 + 1] +
                                 wr[2] * sIn[p * 4 + 2] + b0[j]));
        }
        sActA[i] = (_Float16)v;
        sActB[i] = (_Float16)0.0f;
        sEmb[i]  = (_Float16)0.0f;
    }

    // ---- main layer loop ----
    const int hi  = lane >> 4;
    const int nlo = lane & 15;
    const int waveRow = wave * 16 * KPAD;
    v8f xout[9];
#pragma unroll
    for (int nt = 0; nt < 9; ++nt) xout[nt] = (v8f){};

    for (int l = 1; l < NLAYER; ++l) {
        const _Float16* aIn = (l & 1) ? sActA : sActB;   // ping-pong activations
        _Float16* aOut      = (l & 1) ? sActB : sActA;

        // ---- stage ff_W[l-1]: wait TDM, convert f32->f16, kick next DMA ----
        __syncthreads();                       // all waves done with old sW
        if (wave == 0) tdm_wait();             // sStage holds ff_W[l-1]
        __syncthreads();
#if HAVE_TDM
        const float* srcF = sStage;
#else
        const float* srcF = Wf + (size_t)(l - 1) * DLEN * DLEN;
#endif
        for (int i = tid; i < DLEN * DLEN; i += 256) {
            const int j = i / DLEN, k = i - j * DLEN;
            sW[j * KPAD + k] = (_Float16)srcF[i];
        }
        __syncthreads();
        if (wave == 0) tdm_load_matrix(Wo + (size_t)(l - 1) * DLEN * DLEN, stageOff);

        // ---- matmul 1 (per tile) + elementwise: x=sin(w0*(..)), emb=x+posenc ----
        {
            const float* bfl = bfv + (size_t)(l - 1) * DLEN;
            const int lvl = l - 1;
#pragma unroll
            for (int nt = 0; nt < 9; ++nt) {
                v8f c = mm_tile(aIn + waveRow, sW, nt * 16, lane);
                const int n_e = nt * 16 + nlo;
                const float bb = bfl[(n_e < DLEN) ? n_e : 0] * ((n_e < DLEN) ? 1.0f : 0.0f);
#pragma unroll
                for (int r = 0; r < 8; ++r) {
                    const int m_e = r + hi * 8;           // C layout row
                    const int p   = wave * 16 + m_e;
                    const float xv = __sinf(SIN_W0 * (c[r] + bb));
                    const float pe = posenc_val(&sGrid[p * 64 + lvl * 8], n_e);
                    aOut[p * KPAD + n_e] = (_Float16)xv;
                    sEmb[p * KPAD + n_e] = (_Float16)(xv + pe);
                }
            }
        }

        // ---- stage out_W[l-1] ----
        __syncthreads();
        if (wave == 0) tdm_wait();             // sStage holds out_W[l-1]
        __syncthreads();
#if HAVE_TDM
        const float* srcO = sStage;
#else
        const float* srcO = Wo + (size_t)(l - 1) * DLEN * DLEN;
#endif
        for (int i = tid; i < DLEN * DLEN; i += 256) {
            const int j = i / DLEN, k = i - j * DLEN;
            sW[j * KPAD + k] = (_Float16)srcO[i];
        }
        __syncthreads();
        if (wave == 0 && l < NLAYER - 1)
            tdm_load_matrix(Wf + (size_t)l * DLEN * DLEN, stageOff);

        // ---- matmul 2: x_out += sin(w0h*(emb @ out_W^T + b)) ----
        {
            const float* bol = bov + (size_t)(l - 1) * DLEN;
#pragma unroll
            for (int nt = 0; nt < 9; ++nt) {
                v8f c2 = mm_tile(sEmb + waveRow, sW, nt * 16, lane);
                const int n_e = nt * 16 + nlo;
                const float bb = bol[(n_e < DLEN) ? n_e : 0] * ((n_e < DLEN) ? 1.0f : 0.0f);
#pragma unroll
                for (int r = 0; r < 8; ++r)
                    xout[nt][r] += __sinf(SIN_W0H * (c2[r] + bb));
            }
        }
    }

    // ---- epilogue: out = [xin | x_out/8], stride 139 ----
#pragma unroll
    for (int nt = 0; nt < 9; ++nt) {
        const int n_e = nt * 16 + nlo;
        if (n_e < DLEN) {
#pragma unroll
            for (int r = 0; r < 8; ++r) {
                const int m_e = r + hi * 8;
                const size_t row = (size_t)(blockP + wave * 16 + m_e);
                out[row * 139 + 3 + n_e] = xout[nt][r] * 0.125f;
            }
        }
    }
    if (tid < TPTS) {
        const size_t row = (size_t)(blockP + tid);
#pragma unroll
        for (int c = 0; c < 3; ++c)
            out[row * 139 + c] = (sIn[tid * 4 + c] + 1.0f) * 0.5f;
    }
}

extern "C" void kernel_launch(void* const* d_in, const int* in_sizes, int n_in,
                              void* d_out, int out_size, void* d_ws, size_t ws_size,
                              hipStream_t stream) {
    const float* inp    = (const float*)d_in[0];
    const float* tables = (const float*)d_in[1];
    const float* ff_W0  = (const float*)d_in[2];
    const float* ff_b0  = (const float*)d_in[3];
    const float* ff_W   = (const float*)d_in[4];
    const float* ff_b   = (const float*)d_in[5];
    const float* out_W  = (const float*)d_in[6];
    const float* out_b  = (const float*)d_in[7];
    float* outp = (float*)d_out;

    const int npts   = in_sizes[0] / 3;       // 131072
    const int blocks = npts / TPTS;           // 1024

    hipLaunchKernelGGL(FourierFilterBanks_52115133170147_kernel,
                       dim3(blocks), dim3(256), SMEM_BYTES, stream,
                       inp, tables, ff_W0, ff_b0, ff_W, ff_b, out_W, out_b, outp);
}
